// EnCoder_22797686407620
// MI455X (gfx1250) — matrix operand order repaired and checked by reading, not compiled, not run on hardware
//
#include <hip/hip_runtime.h>
#include <hip/hip_bf16.h>

// ---------------------------------------------------------------------------
// Bidirectional GRU (LFADS-style) for MI455X / gfx1250.
//   Kernel 1: x = inputs @ win_w^T + win_b  -> xw[t][b*N+n]  (tiny, VALU)
//   Kernel 2: the recurrence, bf16 WMMA GEMMs with all weights resident in LDS.
// B=16, T=64, N=512, D=128 -> 8192 independent GRU cells per direction.
// ---------------------------------------------------------------------------

typedef __bf16 bf16_t;
typedef __attribute__((ext_vector_type(16))) __bf16 v16bf;
typedef __attribute__((ext_vector_type(8)))  __bf16 v8bf;
typedef __attribute__((ext_vector_type(8)))  float  v8f;

#define B_   16
#define T_   64
#define N_   512
#define D_   128
#define R_   (B_ * N_)        // 8192 rows (b*N + n)
#define MWG  32               // rows per workgroup
#define WP   136              // padded bf16 row pitch (272 B, 16B-aligned, breaks bank stride)
#define CLIPV 10000.0f

// LDS byte offsets (all multiples of 16)
#define OFF_WRUT 0            // 256*136*2 = 69632   W_ru^T (h-part), bf16
#define OFF_WCT  69632        // 128*136*2 = 34816   W_c^T  (h-part), bf16
#define OFF_HBF  104448       // 32*136*2  = 8704    h as bf16 (A operand GEMM1)
#define OFF_RH   113152       // 32*136*2  = 8704    r*h as bf16 (A operand GEMM2)
#define OFF_H    121856       // 32*128*4  = 16384   h, f32
#define OFF_U    138240       // 32*128*4  = 16384   u, f32
#define OFF_BRU  154624       // 256*4               b_ru
#define OFF_WRU0 155648       // 256*4               W_ru row 0 (x weights)
#define OFF_BC   156672       // 128*4               b_c
#define OFF_WC0  157184       // 128*4               W_c row 0
#define OFF_X    157696       // 32*4                x for this step
#define LDS_BYTES 157824

__device__ __forceinline__ float fast_sigmoid(float x) {
    return __builtin_amdgcn_rcpf(1.0f + __expf(-x));
}
__device__ __forceinline__ float fast_tanh(float x) {
    float e = __expf(2.0f * x);
    return 1.0f - 2.0f * __builtin_amdgcn_rcpf(e + 1.0f);
}
__device__ __forceinline__ v16bf cat16(v8bf lo, v8bf hi) {
    return __builtin_shufflevector(lo, hi, 0,1,2,3,4,5,6,7,8,9,10,11,12,13,14,15);
}
__device__ __forceinline__ v8f wmma_bf16(v16bf a, v16bf b, v8f c) {
    return __builtin_amdgcn_wmma_f32_16x16x32_bf16(false, a, false, b, (short)0, c,
                                                   false, false);
}

// ---------------------------------------------------------------------------
// Kernel 1: input linear. xw[t*R + b*N + n] = sum_k inputs[b,t,k]*win_w[n,k] + win_b[n]
// ---------------------------------------------------------------------------
__global__ __launch_bounds__(256)
void lin_kernel(const float* __restrict__ inputs, const float* __restrict__ win_w,
                const float* __restrict__ win_b, float* __restrict__ xw) {
    __shared__ float sx[N_];
    const int bt  = blockIdx.x;          // b*T + t
    const int b   = bt / T_;
    const int t   = bt % T_;
    const int tid = threadIdx.x;
    sx[tid]       = inputs[(size_t)bt * N_ + tid];
    sx[tid + 256] = inputs[(size_t)bt * N_ + tid + 256];
    __syncthreads();
    #pragma unroll
    for (int rep = 0; rep < 2; ++rep) {
        const int n = tid + rep * 256;
        float acc = win_b[n];
        const float* w = win_w + (size_t)n * N_;
        #pragma unroll 8
        for (int k = 0; k < N_; ++k) acc += sx[k] * w[k];
        xw[(size_t)t * R_ + (size_t)b * N_ + n] = acc;
    }
}

// ---------------------------------------------------------------------------
// Kernel 2: GRU recurrence. 512 blocks x 512 threads (16 wave32).
//   blocks [0,256)  : forward scan
//   blocks [256,512): backward scan
// Each block owns 32 rows; weights live in LDS as bf16, transposed so the
// B-operand (K x N) is contiguous-in-K per output column.
// ---------------------------------------------------------------------------
__global__ __launch_bounds__(512)
void gru_kernel(const float* __restrict__ xw,
                const float* __restrict__ fWru, const float* __restrict__ fbru,
                const float* __restrict__ fWc,  const float* __restrict__ fbc,
                const float* __restrict__ bWru, const float* __restrict__ bbru,
                const float* __restrict__ bWc,  const float* __restrict__ bbc,
                float* __restrict__ out) {
    extern __shared__ __align__(16) char smem[];
    bf16_t* Wrut = (bf16_t*)(smem + OFF_WRUT);
    bf16_t* Wct  = (bf16_t*)(smem + OFF_WCT);
    bf16_t* hbf  = (bf16_t*)(smem + OFF_HBF);
    bf16_t* rhb  = (bf16_t*)(smem + OFF_RH);
    float*  hls  = (float*)(smem + OFF_H);
    float*  uls  = (float*)(smem + OFF_U);
    float*  brul = (float*)(smem + OFF_BRU);
    float*  wru0 = (float*)(smem + OFF_WRU0);
    float*  bcl  = (float*)(smem + OFF_BC);
    float*  wc0  = (float*)(smem + OFF_WC0);
    float*  xls  = (float*)(smem + OFF_X);

    const int tid  = threadIdx.x;
    const int dir  = blockIdx.x >> 8;        // 0 = fwd, 1 = bwd
    const int wg   = blockIdx.x & 255;
    const int base = wg * MWG;

    const float* Wru = dir ? bWru : fWru;
    const float* bru = dir ? bbru : fbru;
    const float* Wc  = dir ? bWc  : fWc;
    const float* bc  = dir ? bbc  : fbc;
    float* outD = out + (size_t)dir * ((size_t)T_ * R_ * D_);

    // ---- one-time: stage weights (transposed, bf16) + biases, zero state ----
    for (int i = tid; i < 128 * 256; i += 512) {      // W_ru rows 1..128 -> Wrut[n][k]
        const int k = i >> 8, n = i & 255;
        Wrut[n * WP + k] = (bf16_t)Wru[(size_t)(k + 1) * 256 + n];
    }
    for (int i = tid; i < 128 * 128; i += 512) {      // W_c rows 1..128 -> Wct[j][k]
        const int k = i >> 7, j = i & 127;
        Wct[j * WP + k] = (bf16_t)Wc[(size_t)(k + 1) * 128 + j];
    }
    if (tid < 256) { brul[tid] = bru[tid]; wru0[tid] = Wru[tid]; }
    if (tid < 128) { bcl[tid]  = bc[tid];  wc0[tid]  = Wc[tid]; }
    for (int i = tid; i < MWG * D_; i += 512) hls[i] = 0.0f;
    for (int i = tid; i < MWG * WP; i += 512) { hbf[i] = (bf16_t)0.0f; rhb[i] = (bf16_t)0.0f; }

    const int wv  = tid >> 5;       // wave 0..15
    const int ln  = tid & 31;
    const int l15 = ln & 15;
    const int hi  = ln >> 4;

    // GEMM1: wave -> row tile mt1 (0..1), column tiles jp, jp+1 of ru (0..15)
    const int mt1 = wv >> 3;
    const int jp  = (wv & 7) * 2;
    // GEMM2: wave -> row tile mt2, column tile jt2 of c (0..7)
    const int mt2 = wv >> 3;
    const int jt2 = wv & 7;

    for (int s = 0; s < T_; ++s) {
        const int t = dir ? (T_ - 1 - s) : s;
        __syncthreads();                                   // prev step's writes visible
        if (tid < MWG) {
            xls[tid] = xw[(size_t)t * R_ + base + tid];
            const int t2 = dir ? (t - 1) : (t + 1);
            if (t2 >= 0 && t2 < T_)
                __builtin_prefetch(xw + (size_t)t2 * R_ + base + tid, 0, 0);
        }
        __syncthreads();

        // ---------------- GEMM1: ru_pre = [x,h] @ W_ru + b_ru ----------------
        v8f acc0, acc1;
        const int n0 = jp * 16 + l15, n1 = n0 + 16;
        {
            const float b0 = brul[n0], w00 = wru0[n0];
            const float b1 = brul[n1], w01 = wru0[n1];
            #pragma unroll
            for (int g = 0; g < 8; ++g) {
                const float xv = xls[mt1 * 16 + hi * 8 + g];
                acc0[g] = b0 + xv * w00;
                acc1[g] = b1 + xv * w01;
            }
            #pragma unroll
            for (int kb = 0; kb < 4; ++kb) {
                const bf16_t* ap = hbf + (mt1 * 16 + l15) * WP + kb * 32 + hi * 16;
                const v16bf a = cat16(*(const v8bf*)ap, *(const v8bf*)(ap + 8));
                const bf16_t* bp0 = Wrut + n0 * WP + kb * 32 + hi * 16;
                const v16bf bm0 = cat16(*(const v8bf*)bp0, *(const v8bf*)(bp0 + 8));
                const bf16_t* bp1 = Wrut + n1 * WP + kb * 32 + hi * 16;
                const v16bf bm1 = cat16(*(const v8bf*)bp1, *(const v8bf*)(bp1 + 8));
                acc0 = wmma_bf16(a, bm0, acc0);
                acc1 = wmma_bf16(a, bm1, acc1);
            }
        }
        if (jp < 8) {               // r tiles: rh = sigmoid(.) * h  -> bf16 LDS
            const int k0 = jp * 16 + l15, k1 = k0 + 16;
            #pragma unroll
            for (int g = 0; g < 8; ++g) {
                const int m = mt1 * 16 + hi * 8 + g;
                rhb[m * WP + k0] = (bf16_t)(fast_sigmoid(acc0[g]) * hls[m * D_ + k0]);
                rhb[m * WP + k1] = (bf16_t)(fast_sigmoid(acc1[g]) * hls[m * D_ + k1]);
            }
        } else {                    // u tiles: u = sigmoid(.)      -> f32 LDS
            const int d0 = (jp - 8) * 16 + l15, d1 = d0 + 16;
            #pragma unroll
            for (int g = 0; g < 8; ++g) {
                const int m = mt1 * 16 + hi * 8 + g;
                uls[m * D_ + d0] = fast_sigmoid(acc0[g]);
                uls[m * D_ + d1] = fast_sigmoid(acc1[g]);
            }
        }
        __syncthreads();

        // ---------------- GEMM2: c_pre = [x, r*h] @ W_c + b_c; combine -------
        v8f accc;
        const int dc = jt2 * 16 + l15;
        {
            const float b0 = bcl[dc], w00 = wc0[dc];
            #pragma unroll
            for (int g = 0; g < 8; ++g) {
                const float xv = xls[mt2 * 16 + hi * 8 + g];
                accc[g] = b0 + xv * w00;
            }
            #pragma unroll
            for (int kb = 0; kb < 4; ++kb) {
                const bf16_t* ap = rhb + (mt2 * 16 + l15) * WP + kb * 32 + hi * 16;
                const v16bf a = cat16(*(const v8bf*)ap, *(const v8bf*)(ap + 8));
                const bf16_t* bp = Wct + dc * WP + kb * 32 + hi * 16;
                const v16bf bm = cat16(*(const v8bf*)bp, *(const v8bf*)(bp + 8));
                accc = wmma_bf16(a, bm, accc);
            }
        }
        #pragma unroll
        for (int g = 0; g < 8; ++g) {
            const int m = mt2 * 16 + hi * 8 + g;
            const float c  = fast_tanh(accc[g]);
            const float u  = uls[m * D_ + dc];
            float hn = u * hls[m * D_ + dc] + (1.0f - u) * c;
            hn = fminf(fmaxf(hn, -CLIPV), CLIPV);
            hls[m * D_ + dc] = hn;
            hbf[m * WP + dc] = (bf16_t)hn;
            outD[((size_t)t * R_ + base + m) * D_ + dc] = hn;
        }
    }
}

// ---------------------------------------------------------------------------
extern "C" void kernel_launch(void* const* d_in, const int* in_sizes, int n_in,
                              void* d_out, int out_size, void* d_ws, size_t ws_size,
                              hipStream_t stream) {
    (void)in_sizes; (void)n_in; (void)out_size; (void)ws_size;
    const float* inputs = (const float*)d_in[0];
    const float* win_w  = (const float*)d_in[1];
    const float* win_b  = (const float*)d_in[2];
    const float* fWru   = (const float*)d_in[3];
    const float* fbru   = (const float*)d_in[4];
    const float* fWc    = (const float*)d_in[5];
    const float* fbc    = (const float*)d_in[6];
    const float* bWru   = (const float*)d_in[7];
    const float* bbru   = (const float*)d_in[8];
    const float* bWc    = (const float*)d_in[9];
    const float* bbc    = (const float*)d_in[10];
    float* out = (float*)d_out;
    float* xw  = (float*)d_ws;   // T*R f32 = 2 MB scratch

    lin_kernel<<<dim3(B_ * T_), dim3(256), 0, stream>>>(inputs, win_w, win_b, xw);
    gru_kernel<<<dim3(512), dim3(512), LDS_BYTES, stream>>>(
        xw, fWru, fbru, fWc, fbc, bWru, bbru, bWc, bbc, out);
}
